// BaseQuadTreeNetwork_55551107006470
// MI455X (gfx1250) — matrix operand order repaired and checked by reading, not compiled, not run on hardware
//
#include <hip/hip_runtime.h>
#include <hip/hip_bf16.h>

typedef __attribute__((ext_vector_type(16))) _Float16 v16h;
typedef __attribute__((ext_vector_type(8)))  _Float16 v8h;
typedef __attribute__((ext_vector_type(8)))  float    v8f;
typedef __attribute__((ext_vector_type(4)))  float    v4f;

#define DEPTH   7
#define M_DIM   128
#define C_DIM   64
#define N_KEYS  21844      // sum 4^i, i=1..7
#define KPAD    21856      // 1366 * 16 (key rows padded to 16-row tiles)
#define NKT     1366       // key tiles
#define KCHUNK  16         // key tiles per wave (query frags reused in regs)
#define BATCH   4096
#define NBT     (BATCH/16) // 256 batch tiles

// ---------------- Stage 0: query = hidden @ Wq^T + bq  -> f16 ----------------
__global__ void query_kernel(const float* __restrict__ hidden,
                             const float* __restrict__ Wq,
                             const float* __restrict__ bq,
                             _Float16* __restrict__ q16) {
    __shared__ float h[C_DIM];
    const int b = blockIdx.x;
    const int m = threadIdx.x;              // 128 threads
    if (m < C_DIM) h[m] = hidden[(size_t)b * C_DIM + m];
    __syncthreads();
    const float* w = Wq + (size_t)m * C_DIM;
    float acc = bq[m];
    #pragma unroll
    for (int c = 0; c < C_DIM; ++c) acc += h[c] * w[c];
    q16[(size_t)b * M_DIM + m] = (_Float16)acc;
}

// ------- Stage 1: keys = relu(states @ Wk[d]^T + bk[d]) -> f16, zero-pad -----
__global__ void keys_kernel(const float* __restrict__ states,
                            const float* __restrict__ Wk,
                            const float* __restrict__ bk,
                            _Float16* __restrict__ k16) {
    __shared__ float s[M_DIM];
    const int r = blockIdx.x;               // key row (padded range)
    const int m = threadIdx.x;              // 128 threads
    if (r >= N_KEYS) {                      // uniform per block
        k16[(size_t)r * M_DIM + m] = (_Float16)0.0f;
        return;
    }
    s[m] = states[(size_t)r * M_DIM + m];
    __syncthreads();
    // find depth: row r belongs to depth d with start = (4^(d+1)-4)/3
    int start = 0, d = 0, n = 4;
    while (r >= start + n) { start += n; n <<= 2; ++d; }
    const float* w = Wk + ((size_t)d * M_DIM + m) * M_DIM;
    float acc = bk[(size_t)d * M_DIM + m];
    #pragma unroll 8
    for (int k = 0; k < M_DIM; ++k) acc += s[k] * w[k];
    acc = acc > 0.0f ? acc : 0.0f;          // relu
    k16[(size_t)r * M_DIM + m] = (_Float16)acc;
}

// ---- Stage 2: scores via WMMA f16->f32, fused per-lane 4-way log-softmax ----
// One wave computes 16-key x 16-batch tiles; query fragments held in VGPRs
// across a chunk of KCHUNK key tiles. logp is written with non-temporal
// stores (TH=NT) so the 358MB stream does not evict the resident f16 keys
// (5.6MB) / query (1MB) operands from the 192MB L2.
__launch_bounds__(256)
__global__ void scores_kernel(const _Float16* __restrict__ k16,
                              const _Float16* __restrict__ q16,
                              float* __restrict__ logp) {
    const int wave = threadIdx.x >> 5;
    const int lane = threadIdx.x & 31;
    const int gw   = blockIdx.x * 8 + wave;       // global wave id
    const int tb   = gw % NBT;                    // batch tile
    const int ch   = gw / NBT;                    // key-tile chunk
    const int t0   = ch * KCHUNK;
    if (t0 >= NKT) return;                        // uniform per wave
    const int t1   = (t0 + KCHUNK < NKT) ? (t0 + KCHUNK) : NKT;

    const int col = lane & 15;                    // batch col / key row
    const int khi = lane >> 4;                    // lane-half selector

    // B fragments (query^T, 32x16 per K-block): lane holds column `col`,
    // K = kk*32 + khi*16 + h  for h=0..15 (contiguous 16 halves).
    v16h bf[4];
    {
        const _Float16* qrow = q16 + (size_t)(tb * 16 + col) * M_DIM;
        #pragma unroll
        for (int kk = 0; kk < 4; ++kk) {
            const _Float16* p = qrow + kk * 32 + khi * 16;
            v8h lo = *(const v8h*)(p);
            v8h hi = *(const v8h*)(p + 8);
            v16h f;
            #pragma unroll
            for (int i = 0; i < 8; ++i) { f[i] = lo[i]; f[i + 8] = hi[i]; }
            bf[kk] = f;
        }
    }

    for (int t = t0; t < t1; ++t) {
        // A fragments (keys, 16x32 per K-block): lane holds row (lane&15),
        // halves 0..15 -> K = kk*32 + khi*16 + h (contiguous 16 halves)
        const _Float16* krow = k16 + (size_t)(t * 16 + col) * M_DIM;
        v8f c = {};
        #pragma unroll
        for (int kk = 0; kk < 4; ++kk) {
            const _Float16* p = krow + kk * 32 + khi * 16;
            v8h lo = *(const v8h*)(p);
            v8h hi = *(const v8h*)(p + 8);
            v16h a;
            #pragma unroll
            for (int i = 0; i < 8; ++i) { a[i] = lo[i]; a[i + 8] = hi[i]; }
            c = __builtin_amdgcn_wmma_f32_16x16x32_f16(
                    false, a, false, bf[kk], (short)0, c, false, false);
        }

        // C/D layout: this lane holds keys (t*16 + khi*8 + e), e=0..7, for
        // batch (tb*16 + col). Groups of 4 consecutive keys = one internal
        // node => per-lane log-softmax over elements {0..3} and {4..7}.
        float lp[8];
        #pragma unroll
        for (int g = 0; g < 2; ++g) {
            float x0 = c[g * 4 + 0], x1 = c[g * 4 + 1];
            float x2 = c[g * 4 + 2], x3 = c[g * 4 + 3];
            float mx = fmaxf(fmaxf(x0, x1), fmaxf(x2, x3));
            float s  = __expf(x0 - mx) + __expf(x1 - mx)
                     + __expf(x2 - mx) + __expf(x3 - mx);
            float ls = mx + __logf(s);
            lp[g * 4 + 0] = x0 - ls; lp[g * 4 + 1] = x1 - ls;
            lp[g * 4 + 2] = x2 - ls; lp[g * 4 + 3] = x3 - ls;
        }
        float* o = logp + (size_t)(tb * 16 + col) * KPAD + t * 16 + khi * 8;
        v4f w0 = { lp[0], lp[1], lp[2], lp[3] };
        v4f w1 = { lp[4], lp[5], lp[6], lp[7] };
        __builtin_nontemporal_store(w0, (v4f*)(o));        // TH=NT b128
        __builtin_nontemporal_store(w1, (v4f*)(o + 4));    // TH=NT b128
    }
}

// ------ Stage 3: cumulative tree propagation, one block per batch elem -------
// logp is read once (last use) and dists written once -> non-temporal on both.
__global__ void tree_kernel(const float* __restrict__ logp,
                            float* __restrict__ out) {
    __shared__ float bufA[4096];
    __shared__ float bufB[4096];
    const int b = blockIdx.x;
    const float* lrow = logp + (size_t)b * KPAD;
    const int keyoff[8] = {0, 0, 4, 20, 84, 340, 1364, 5460}; // (4^d-4)/3

    float* pv = bufA;   // dist at depth d-1
    float* nx = bufB;
    int n = 4;
    size_t ooff = 0;
    for (int d = 1; d <= DEPTH; ++d) {
        const float* lp = lrow + keyoff[d];
        float* o = out + ooff + (size_t)b * n;
        for (int j = threadIdx.x; j < n; j += blockDim.x) {
            float v = __builtin_nontemporal_load(lp + j);  // TH=NT/LU stream
            if (d > 1) v += pv[j >> 2];
            __builtin_nontemporal_store(v, o + j);         // TH=NT stream
            if (d < DEPTH) nx[j] = v;   // max n at d<7 is 4096 -> fits
        }
        __syncthreads();
        float* tmp = pv; pv = nx; nx = tmp;
        ooff += (size_t)BATCH * n;
        n <<= 2;
    }
}

extern "C" void kernel_launch(void* const* d_in, const int* in_sizes, int n_in,
                              void* d_out, int out_size, void* d_ws, size_t ws_size,
                              hipStream_t stream) {
    const float* hidden = (const float*)d_in[0];   // [4096, 64]
    const float* Wq     = (const float*)d_in[1];   // [128, 64]
    const float* bq     = (const float*)d_in[2];   // [128]
    const float* states = (const float*)d_in[3];   // [21844, 128]
    const float* Wk     = (const float*)d_in[4];   // [7, 128, 128]
    const float* bk     = (const float*)d_in[5];   // [7, 128]
    float* out = (float*)d_out;                    // concat dists, 4096*21844

    // workspace layout (sections naturally aligned)
    char* ws = (char*)d_ws;
    float*    logp = (float*)ws;                                   // 4096*21856*4
    _Float16* q16  = (_Float16*)(ws + (size_t)BATCH * KPAD * 4);   // 4096*128*2
    _Float16* k16  = (_Float16*)(ws + (size_t)BATCH * KPAD * 4
                                    + (size_t)BATCH * M_DIM * 2);  // 21856*128*2

    query_kernel<<<BATCH, M_DIM, 0, stream>>>(hidden, Wq, bq, q16);
    keys_kernel <<<KPAD,  M_DIM, 0, stream>>>(states, Wk, bk, k16);

    const int nchunks = (NKT + KCHUNK - 1) / KCHUNK;   // 86
    const int nwaves  = NBT * nchunks;                 // 22016
    const int nblocks = (nwaves + 7) / 8;              // 8 waves/block
    scores_kernel<<<nblocks, 256, 0, stream>>>(k16, q16, logp);

    tree_kernel<<<BATCH, 256, 0, stream>>>(logp, out);
}
